// NnmfCnn_78537771975272
// MI455X (gfx1250) — matrix operand order repaired and checked by reading, not compiled, not run on hardware
//
#include <hip/hip_runtime.h>
#include <hip/hip_bf16.h>

typedef __attribute__((ext_vector_type(16))) _Float16 v16h;
typedef __attribute__((ext_vector_type(8)))  _Float16 v8h;
typedef __attribute__((ext_vector_type(8)))  float    v8f;

#define EPSF   1e-20f
#define BNEPS  1e-5f

// epilogue modes for conv3x3_wmma
#define CMODE_PLAIN 0   // out = acc
#define CMODE_MUL   1   // out = aux * acc          (multiplicative h update)
#define CMODE_DIV   2   // out = aux / (acc + eps)  (fused recon division)

static __device__ __forceinline__ v16h combine16(v8h a, v8h b) {
    return __builtin_shufflevector(a, b, 0,1,2,3,4,5,6,7,8,9,10,11,12,13,14,15);
}

// Async global->LDS copy of one 16-byte chunk per active lane (ASYNCcnt).
static __device__ __forceinline__ void async_copy_b128(
    unsigned lds_byte_addr, const _Float16* gsrc)
{
    asm volatile("global_load_async_to_lds_b128 %0, %1, off"
                 :: "v"(lds_byte_addr), "v"(gsrc) : "memory");
}
static __device__ __forceinline__ void async_wait0() {
    asm volatile("s_wait_asynccnt 0" ::: "memory");
}

// ---------------------------------------------------------------------------
// Generic 3x3 "same" cross-correlation conv as implicit GEMM on WMMA.
//   acc[b,co,y,x] = sum_ci sum_{ky,kx} in[b,ci,y+ky-1,x+kx-1] * wgt[co,ci,ky,kx]
// Tiling: block = 256 threads = 8 waves; tile = 16 co x (8 rows x 32 cols).
//         Each wave: one row, two 16-col sub-tiles sharing the A fragment.
// LDS input tile is pixel-major [10 rows][34 cols][ci stride 40 halves] so
// B fragments load as contiguous 16-byte chunks (ds_load_b128).
// Weights are pre-packed f16 tiles [coBlk][ciBlk][16][9][32]; each chunk's
// tile is a contiguous 9216B region staged with global_load_async_to_lds.
// ---------------------------------------------------------------------------
__global__ __launch_bounds__(256) void conv3x3_wmma(
    const float* __restrict__ in,        // [NB, CI, H, W]
    const _Float16* __restrict__ wgt16,  // packed tiles [coB][ciB][16][9][32]
    const float* __restrict__ aux,       // epilogue operand or nullptr
    float* __restrict__ out,             // [NB, CO, H, W]
    int CI, int CO, int H, int W, int nCiB, int mode)
{
    __shared__ _Float16 lds_in[10 * 34 * 40];   // pixel-major, padded ci stride
    __shared__ _Float16 lds_w [16 * 9 * 32];    // [co][tap][ci]

    const int tid  = threadIdx.x;
    const int lane = tid & 31;
    const int wv   = tid >> 5;          // wave id = local row 0..7
    const int n    = lane & 15;         // matrix column within sub-tile
    const int hi   = lane >> 4;         // lane half

    const int tiles_x = W >> 5;
    const int tx = blockIdx.x % tiles_x;
    const int ty = blockIdx.x / tiles_x;
    const int x0 = tx * 32;
    const int y0 = ty * 8;
    const int co0 = blockIdx.y * 16;
    const int b   = blockIdx.z;

    v8f acc0 = {0.f,0.f,0.f,0.f,0.f,0.f,0.f,0.f};
    v8f acc1 = {0.f,0.f,0.f,0.f,0.f,0.f,0.f,0.f};

    const int abase = hi * 16;
    const int bbase = hi * 16;
    const unsigned ldsw_base = (unsigned)(unsigned long long)&lds_w[0];

    for (int ci0 = 0; ci0 < CI; ci0 += 32) {
        // ---- async-stage packed weight tile (contiguous 9216 B) ----
        {
            const _Float16* wsrc =
                wgt16 + ((size_t)(co0 >> 4) * nCiB + (ci0 >> 5)) * (16 * 9 * 32);
            for (int idx = tid; idx < 576; idx += 256)   // 576 x 16B
                async_copy_b128(ldsw_base + idx * 16, wsrc + idx * 8);
        }
        // ---- stage input tile (halo included), x-coalesced reads ----
        for (int idx = tid; idx < 32 * 10 * 34; idx += 256) {
            int ci  = idx / 340;
            int rem = idx - ci * 340;
            int r   = rem / 34;
            int c   = rem - r * 34;
            int y   = y0 - 1 + r;
            int x   = x0 - 1 + c;
            int cig = ci0 + ci;
            float v = 0.f;
            if (cig < CI && y >= 0 && y < H && x >= 0 && x < W)
                v = in[(((size_t)b * CI + cig) * H + y) * W + x];
            lds_in[(r * 34 + c) * 40 + ci] = (_Float16)v;
        }
        async_wait0();
        __syncthreads();

        #pragma unroll
        for (int ky = 0; ky < 3; ++ky) {
            #pragma unroll
            for (int kx = 0; kx < 3; ++kx) {
                const int tap = ky * 3 + kx;
                // A (16x32 f16): halves {base..base+15}
                const _Float16* ap = &lds_w[n * 288 + tap * 32 + abase];
                v16h A = combine16(*(const v8h*)ap, *(const v8h*)(ap + 8));
                // B sub-tile 0: pixel (wv+ky, n+kx)
                const _Float16* bp0 =
                    &lds_in[((wv + ky) * 34 + (n + kx)) * 40 + bbase];
                v16h B0 = combine16(*(const v8h*)bp0, *(const v8h*)(bp0 + 8));
                acc0 = __builtin_amdgcn_wmma_f32_16x16x32_f16(
                           false, A, false, B0, (short)0, acc0, false, false);
                // B sub-tile 1: pixel (wv+ky, 16+n+kx)
                const _Float16* bp1 =
                    &lds_in[((wv + ky) * 34 + (16 + n + kx)) * 40 + bbase];
                v16h B1 = combine16(*(const v8h*)bp1, *(const v8h*)(bp1 + 8));
                acc1 = __builtin_amdgcn_wmma_f32_16x16x32_f16(
                           false, A, false, B1, (short)0, acc1, false, false);
            }
        }
        __syncthreads();
    }

    // D layout: reg r holds M = r + hi*8, column n.
    const int y = y0 + wv;
    #pragma unroll
    for (int r = 0; r < 8; ++r) {
        int co = co0 + r + hi * 8;
        if (co < CO) {
            size_t row = (((size_t)b * CO + co) * H + y) * W;
            size_t i0 = row + x0 + n;
            size_t i1 = row + x0 + 16 + n;
            float v0 = acc0[r], v1 = acc1[r];
            if (mode == CMODE_MUL)      { v0 *= aux[i0];               v1 *= aux[i1]; }
            else if (mode == CMODE_DIV) { v0 = aux[i0] / (v0 + EPSF);  v1 = aux[i1] / (v1 + EPSF); }
            out[i0] = v0;
            out[i1] = v1;
        }
    }
}

// ---------------------------------------------------------------------------
// 1x1 conv + bias + ReLU as WMMA GEMM:  out = relu(W*h + b)
// Same tiling: 16 co x (8 rows x 32 cols) per block; 2 sub-tiles per wave.
// Weights pre-packed f16 tiles [coB][ciB][16][32] (1024 B contiguous).
// ---------------------------------------------------------------------------
__global__ __launch_bounds__(256) void gemm1x1_relu_wmma(
    const float* __restrict__ in,        // [NB, CI, H, W]
    const _Float16* __restrict__ wgt16,  // packed tiles [coB][ciB][16][32]
    const float* __restrict__ bias,      // [CO]
    float* __restrict__ out,             // [NB, CO, H, W]
    int CI, int CO, int H, int W, int nCiB)
{
    __shared__ _Float16 lds_in[8 * 32 * 40];   // pixel-major, padded ci stride
    __shared__ _Float16 lds_w [16 * 32];       // [co][ci]

    const int tid  = threadIdx.x;
    const int lane = tid & 31;
    const int wv   = tid >> 5;
    const int n    = lane & 15;
    const int hi   = lane >> 4;

    const int tiles_x = W >> 5;
    const int tx = blockIdx.x % tiles_x;
    const int ty = blockIdx.x / tiles_x;
    const int x0 = tx * 32;
    const int y0 = ty * 8;
    const int co0 = blockIdx.y * 16;
    const int b   = blockIdx.z;

    v8f acc0 = {0.f,0.f,0.f,0.f,0.f,0.f,0.f,0.f};
    v8f acc1 = {0.f,0.f,0.f,0.f,0.f,0.f,0.f,0.f};

    const int abase = hi * 16;
    const int bbase = hi * 16;
    const unsigned ldsw_base = (unsigned)(unsigned long long)&lds_w[0];

    for (int ci0 = 0; ci0 < CI; ci0 += 32) {
        {
            const _Float16* wsrc =
                wgt16 + ((size_t)(co0 >> 4) * nCiB + (ci0 >> 5)) * (16 * 32);
            if (tid < 64)                                 // 64 x 16B = 1 KB
                async_copy_b128(ldsw_base + tid * 16, wsrc + tid * 8);
        }
        for (int idx = tid; idx < 32 * 8 * 32; idx += 256) {
            int ci  = idx >> 8;             // /256
            int rem = idx & 255;
            int r   = rem >> 5;
            int c   = rem & 31;
            int cig = ci0 + ci;
            float v = 0.f;
            if (cig < CI)
                v = in[(((size_t)b * CI + cig) * H + y0 + r) * W + x0 + c];
            lds_in[(r * 32 + c) * 40 + ci] = (_Float16)v;
        }
        async_wait0();
        __syncthreads();

        const _Float16* ap = &lds_w[n * 32 + abase];
        v16h A = combine16(*(const v8h*)ap, *(const v8h*)(ap + 8));
        const _Float16* bp0 = &lds_in[(wv * 32 + n) * 40 + bbase];
        v16h B0 = combine16(*(const v8h*)bp0, *(const v8h*)(bp0 + 8));
        acc0 = __builtin_amdgcn_wmma_f32_16x16x32_f16(
                   false, A, false, B0, (short)0, acc0, false, false);
        const _Float16* bp1 = &lds_in[(wv * 32 + 16 + n) * 40 + bbase];
        v16h B1 = combine16(*(const v8h*)bp1, *(const v8h*)(bp1 + 8));
        acc1 = __builtin_amdgcn_wmma_f32_16x16x32_f16(
                   false, A, false, B1, (short)0, acc1, false, false);
        __syncthreads();
    }

    const int y = y0 + wv;
    #pragma unroll
    for (int r = 0; r < 8; ++r) {
        int co = co0 + r + hi * 8;
        if (co < CO) {
            size_t row = (((size_t)b * CO + co) * H + y) * W;
            float bv = bias[co];
            float v0 = acc0[r] + bv; v0 = v0 > 0.f ? v0 : 0.f;
            float v1 = acc1[r] + bv; v1 = v1 > 0.f ? v1 : 0.f;
            out[row + x0 + n]      = v0;
            out[row + x0 + 16 + n] = v1;
        }
    }
}

// ---------------------------------------------------------------------------
// Weight prep / packing
// ---------------------------------------------------------------------------
__global__ __launch_bounds__(256) void prep_wn(
    const float* __restrict__ w, float* __restrict__ wn, int CIK)
{
    const int co = blockIdx.x;
    const int tid = threadIdx.x;
    float s = 0.f;
    for (int i = tid; i < CIK; i += 256) s += fabsf(w[(size_t)co * CIK + i]);
    __shared__ float red[256];
    red[tid] = s; __syncthreads();
    for (int st = 128; st > 0; st >>= 1) {
        if (tid < st) red[tid] += red[tid + st];
        __syncthreads();
    }
    const float inv = 1.f / (red[0] + EPSF);
    for (int i = tid; i < CIK; i += 256)
        wn[(size_t)co * CIK + i] = fabsf(w[(size_t)co * CIK + i]) * inv;
}

// W2[ci][co][ky][kx] = wn[co][ci][2-ky][2-kx]
__global__ __launch_bounds__(256) void prep_w2(
    const float* __restrict__ wn, float* __restrict__ w2, int CO, int CI)
{
    const size_t total = (size_t)CO * CI * 9;
    for (size_t i = (size_t)blockIdx.x * 256 + threadIdx.x; i < total;
         i += (size_t)gridDim.x * 256) {
        int co  = (int)(i / ((size_t)CI * 9));
        int rem = (int)(i - (size_t)co * CI * 9);
        int ci  = rem / 9;
        int t   = rem - ci * 9;
        int ky  = t / 3, kx = t - ky * 3;
        w2[((size_t)ci * CO + co) * 9 + (2 - ky) * 3 + (2 - kx)] = wn[i];
    }
}

// pack f32 [CO][CI][ntap] -> f16 tiles [coB][ciB][16][ntap][32], zero padded
__global__ __launch_bounds__(256) void pack_w16(
    const float* __restrict__ w, _Float16* __restrict__ out,
    int CO, int CI, int ntap)
{
    const int nCiB = (CI + 31) >> 5;
    const int nCoB = (CO + 15) >> 4;
    const int tile = 16 * ntap * 32;
    const size_t total = (size_t)nCoB * nCiB * tile;
    for (size_t i = (size_t)blockIdx.x * 256 + threadIdx.x; i < total;
         i += (size_t)gridDim.x * 256) {
        size_t t = i / tile;
        int r    = (int)(i - t * tile);
        int cob  = (int)(t / nCiB);
        int cib  = (int)(t - (size_t)cob * nCiB);
        int co_l = r / (ntap * 32);
        int r2   = r - co_l * (ntap * 32);
        int tap  = r2 >> 5;
        int ci_l = r2 & 31;
        int co = cob * 16 + co_l;
        int ci = cib * 32 + ci_l;
        float v = (co < CO && ci < CI)
                ? w[((size_t)co * CI + ci) * ntap + tap] : 0.f;
        out[i] = (_Float16)v;
    }
}

// ---------------------------------------------------------------------------
// Elementwise / reduction helpers
// ---------------------------------------------------------------------------
__global__ __launch_bounds__(256) void relu_l1norm(
    const float* __restrict__ x, float* __restrict__ out,
    int NB, int C, int H, int W)
{
    const size_t plane = (size_t)H * W;
    const size_t total = (size_t)NB * plane;
    for (size_t i = (size_t)blockIdx.x * 256 + threadIdx.x; i < total;
         i += (size_t)gridDim.x * 256) {
        size_t b = i / plane, p = i - b * plane;
        float s = 0.f;
        for (int c = 0; c < C; ++c) {
            float v = x[((size_t)b * C + c) * plane + p];
            s += v > 0.f ? v : 0.f;
        }
        float inv = 1.f / (s + EPSF);
        for (int c = 0; c < C; ++c) {
            float v = x[((size_t)b * C + c) * plane + p];
            out[((size_t)b * C + c) * plane + p] = (v > 0.f ? v : 0.f) * inv;
        }
    }
}

__global__ __launch_bounds__(256) void norm_channels(
    float* __restrict__ h, int NB, int C, int H, int W)
{
    const size_t plane = (size_t)H * W;
    const size_t total = (size_t)NB * plane;
    for (size_t i = (size_t)blockIdx.x * 256 + threadIdx.x; i < total;
         i += (size_t)gridDim.x * 256) {
        size_t b = i / plane, p = i - b * plane;
        float s = 0.f;
        for (int c = 0; c < C; ++c) s += h[((size_t)b * C + c) * plane + p];
        float inv = 1.f / (s + EPSF);
        for (int c = 0; c < C; ++c) h[((size_t)b * C + c) * plane + p] *= inv;
    }
}

__global__ __launch_bounds__(256) void fill_const(
    float* __restrict__ p, size_t n, float v)
{
    for (size_t i = (size_t)blockIdx.x * 256 + threadIdx.x; i < n;
         i += (size_t)gridDim.x * 256) p[i] = v;
}

__global__ __launch_bounds__(256) void bn_stats(
    const float* __restrict__ y, float* __restrict__ stats,
    int NB, int C, int H, int W)
{
    const int c = blockIdx.x;
    const int tid = threadIdx.x;
    const size_t plane = (size_t)H * W;
    const size_t cnt = (size_t)NB * plane;
    float s = 0.f, sq = 0.f;
    for (size_t i = tid; i < cnt; i += 256) {
        size_t b = i / plane, p = i - b * plane;
        float v = y[((size_t)b * C + c) * plane + p];
        s += v; sq += v * v;
    }
    __shared__ float r1[256], r2[256];
    r1[tid] = s; r2[tid] = sq; __syncthreads();
    for (int st = 128; st > 0; st >>= 1) {
        if (tid < st) { r1[tid] += r1[tid + st]; r2[tid] += r2[tid + st]; }
        __syncthreads();
    }
    if (tid == 0) {
        float m = r1[0] / (float)cnt;
        stats[c]     = m;
        stats[C + c] = r2[0] / (float)cnt - m * m;
    }
}

__global__ __launch_bounds__(256) void bn_pool(
    const float* __restrict__ y, const float* __restrict__ stats,
    const float* __restrict__ gamma, const float* __restrict__ beta,
    float* __restrict__ out, int NB, int C, int H, int W)
{
    const int Ho = H >> 1, Wo = W >> 1;
    const size_t total = (size_t)NB * C * Ho * Wo;
    for (size_t i = (size_t)blockIdx.x * 256 + threadIdx.x; i < total;
         i += (size_t)gridDim.x * 256) {
        int xo = (int)(i % Wo);
        size_t t = i / Wo;
        int yo = (int)(t % Ho); t /= Ho;
        int c  = (int)(t % C);
        int b  = (int)(t / C);
        float m   = stats[c];
        float inv = rsqrtf(stats[C + c] + BNEPS);
        float g = gamma[c], be = beta[c];
        const float* base = y + (((size_t)b * C + c) * H + 2 * yo) * W + 2 * xo;
        float s = 0.f;
        s += g * (base[0]     - m) * inv + be;
        s += g * (base[1]     - m) * inv + be;
        s += g * (base[W]     - m) * inv + be;
        s += g * (base[W + 1] - m) * inv + be;
        out[i] = 0.25f * s;
    }
}

// ---------------------------------------------------------------------------
// Host orchestration
// ---------------------------------------------------------------------------
static inline unsigned gblocks(size_t n) {
    size_t b = (n + 255) / 256;
    return (unsigned)(b > 1048576 ? 1048576 : b);
}

extern "C" void kernel_launch(void* const* d_in, const int* in_sizes, int n_in,
                              void* d_out, int out_size, void* d_ws, size_t ws_size,
                              hipStream_t stream) {
    (void)in_sizes; (void)n_in; (void)out_size; (void)ws_size;
    const int FEAT[4] = {3, 64, 128, 256};
    const int NB = 16;

    char* ws = (char*)d_ws;
    const size_t MB = 1ull << 20;
    float* XN = (float*)(ws + 0 * MB);     // <= 16 MB  normalized block input
    float* Hh = (float*)(ws + 16 * MB);    // <= 64 MB  h
    float* R  = (float*)(ws + 80 * MB);    // <= 16 MB  t = x/(recon+eps)
    float* Y  = (float*)(ws + 96 * MB);    // <= 64 MB  relu(1x1 conv) output
    float* P  = (float*)(ws + 160 * MB);   // <= 16 MB  pooled (next input)
    float* WN = (float*)(ws + 176 * MB);   // <= 2 MB   normalized weights (f32)
    float* W2 = (float*)(ws + 178 * MB);   // <= 2 MB   flipped/transposed (f32)
    _Float16* WN16 = (_Float16*)(ws + 180 * MB);  // <= 1 MB packed f16 tiles
    _Float16* W216 = (_Float16*)(ws + 181 * MB);  // <= 1 MB packed f16 tiles
    _Float16* WC16 = (_Float16*)(ws + 182 * MB);  // <= 1 MB packed 1x1 tiles
    float* ST = (float*)(ws + 183 * MB);          // BN stats [2*C]

    const float* x_in = (const float*)d_in[0];
    int Hdim = 128;

    for (int bk = 0; bk < 3; ++bk) {
        const int CI = FEAT[bk], CO = FEAT[bk + 1];
        const int H = Hdim, W = Hdim;
        const float* xin   = (bk == 0) ? x_in : P;
        const float* wnnmf = (const float*)d_in[1 + 5 * bk];
        const float* wconv = (const float*)d_in[2 + 5 * bk];
        const float* bconv = (const float*)d_in[3 + 5 * bk];
        const float* gamma = (const float*)d_in[4 + 5 * bk];
        const float* beta  = (const float*)d_in[5 + 5 * bk];
        const int CIK = CI * 9;
        const int nCiB_f = (CI + 31) / 32;   // fwd conv contracts over CI
        const int nCiB_r = (CO + 31) / 32;   // recon conv contracts over CO
        const int nCiB_g = (CO + 31) / 32;   // 1x1 gemm contracts over CO

        // weight prep: wn = |w| / L1(per filter);  W2 = flip+transpose(wn)
        prep_wn<<<CO, 256, 0, stream>>>(wnnmf, WN, CIK);
        prep_w2<<<gblocks((size_t)CO * CI * 9), 256, 0, stream>>>(WN, W2, CO, CI);
        // pack to f16 tile layouts for async LDS staging
        pack_w16<<<gblocks((size_t)((CO + 15) / 16) * nCiB_f * 4608), 256, 0,
                   stream>>>(WN, WN16, CO, CI, 9);
        pack_w16<<<gblocks((size_t)((CI + 15) / 16) * nCiB_r * 4608), 256, 0,
                   stream>>>(W2, W216, CI, CO, 9);
        pack_w16<<<gblocks((size_t)((CO + 15) / 16) * nCiB_g * 512), 256, 0,
                   stream>>>(wconv, WC16, CO, CO, 1);

        // x = relu(x) / L1_channels
        relu_l1norm<<<gblocks((size_t)NB * H * W), 256, 0, stream>>>(
            xin, XN, NB, CI, H, W);

        // h init uniform
        const size_t hcnt = (size_t)NB * CO * H * W;
        fill_const<<<gblocks(hcnt), 256, 0, stream>>>(Hh, hcnt, 1.f / (float)CO);

        dim3 gR((W / 32) * (H / 8), (unsigned)((CI + 15) / 16), (unsigned)NB);
        dim3 gU((W / 32) * (H / 8), (unsigned)((CO + 15) / 16), (unsigned)NB);

        for (int it = 0; it < 5; ++it) {
            // t = x / (conv(h, W2) + eps)  — division fused into conv epilogue
            conv3x3_wmma<<<gR, 256, 0, stream>>>(Hh, W216, XN, R,
                                                 CO, CI, H, W, nCiB_r, CMODE_DIV);
            // h = h * conv(t, wn)          — multiply fused into conv epilogue
            conv3x3_wmma<<<gU, 256, 0, stream>>>(R, WN16, Hh, Hh,
                                                 CI, CO, H, W, nCiB_f, CMODE_MUL);
            // h /= L1_channels(h)
            norm_channels<<<gblocks((size_t)NB * H * W), 256, 0, stream>>>(
                Hh, NB, CO, H, W);
        }

        // y = relu(W1x1 * h + b)
        gemm1x1_relu_wmma<<<gU, 256, 0, stream>>>(Hh, WC16, bconv, Y,
                                                  CO, CO, H, W, nCiB_g);
        // BN stats (batch mean/var per channel)
        bn_stats<<<CO, 256, 0, stream>>>(Y, ST, NB, CO, H, W);
        // BN apply + 2x2 avgpool
        float* pooled = (bk == 2) ? (float*)d_out : P;
        bn_pool<<<gblocks((size_t)NB * CO * (H / 2) * (W / 2)), 256, 0, stream>>>(
            Y, ST, gamma, beta, pooled, NB, CO, H, W);

        Hdim >>= 1;
    }
}
